// GraphClassifier_30734785970360
// MI455X (gfx1250) — matrix operand order repaired and checked by reading, not compiled, not run on hardware
//
#include <hip/hip_runtime.h>
#include <hip/hip_bf16.h>

// ---------------------------------------------------------------------------
// Types for CDNA5 WMMA (wave32, 16x16x32 bf16 -> f32)
// ---------------------------------------------------------------------------
typedef __attribute__((ext_vector_type(16))) __bf16          v16bf;
typedef __attribute__((ext_vector_type(2)))  __bf16          v2bf;
typedef __attribute__((ext_vector_type(16))) unsigned short  vus16;
typedef __attribute__((ext_vector_type(8)))  float           v8f;

union Frag16 { vus16 u; v16bf b; };

// Pair convert: clang lowers adjacent casts to v_cvt_pk_bf16_f32 where available.
__device__ __forceinline__ unsigned cvt2(float x, float y) {
  v2bf r;
  r.x = (__bf16)x;
  r.y = (__bf16)y;
  return __builtin_bit_cast(unsigned, r);
}

__device__ __forceinline__ unsigned short cvt1(float x) {
  return __builtin_bit_cast(unsigned short, (__bf16)x);
}

// ---------------------------------------------------------------------------
// Degree / norm kernels
// ---------------------------------------------------------------------------
__global__ void init_deg_k(float* deg, int n) {
  int i = blockIdx.x * blockDim.x + threadIdx.x;
  if (i < n) deg[i] = 1.0f;                      // self-loop weight
}

__global__ void deg_accum_k(const int* __restrict__ col,
                            const float* __restrict__ ew,
                            float* deg, int e) {
  int i = blockIdx.x * blockDim.x + threadIdx.x;
  if (i < e) atomicAdd(&deg[col[i]], ew[i]);
}

__global__ void dis_k(const float* __restrict__ deg, float* dis, int n) {
  int i = blockIdx.x * blockDim.x + threadIdx.x;
  if (i < n) {
    float d = deg[i];
    dis[i] = (d > 0.0f) ? rsqrtf(fmaxf(d, 1e-30f)) : 0.0f;
  }
}

// ---------------------------------------------------------------------------
// Pack 128x128 fp32 weight into bf16 B-fragment order (ISA 7.12.2):
// Wp[nt][kk][lane][j]; col=nt*16+(lane&15), half=lane>>4,
//   K=kk*32+half*8+j for j<8 ; K=kk*32+half*8+8+j for j>=8
// ---------------------------------------------------------------------------
__global__ void pack_w_k(const float* __restrict__ W, unsigned short* __restrict__ Wp) {
  int t = blockIdx.x * blockDim.x + threadIdx.x;   // 0..16383
  if (t >= 8 * 4 * 32 * 16) return;
  int j    = t & 15;
  int lane = (t >> 4) & 31;
  int kk   = (t >> 9) & 3;
  int nt   = t >> 11;
  int col  = nt * 16 + (lane & 15);
  int k    = kk * 32 + ((lane >> 4) << 3) + ((j < 8) ? j : (8 + j));
  Wp[t] = cvt1(W[k * 128 + col]);
}

// ---------------------------------------------------------------------------
// Pack activations [N x 128] fp32 -> bf16 in A-fragment order (per 16-row tile):
// Apk[tile][kk][lane][j]; row = tile*16 + (lane&15), half = lane>>4,
//   k0 = kk*32 + half*8, element j<8 -> K=k0+j ; j>=8 -> K=k0+8+j (i.e. k0+16..23)
// Optional fused ReLU (conv-2 input). One thread converts a (j,j+1) pair,
// which is K-contiguous, so loads are float2 and stores are packed dwords.
// ---------------------------------------------------------------------------
__global__ void pack_a_k(const float* __restrict__ X,
                         unsigned* __restrict__ Ap2,
                         int do_relu, long npairs) {
  long t = (long)blockIdx.x * blockDim.x + threadIdx.x;
  if (t >= npairs) return;
  long p   = t * 2;                 // even element index
  int  j   = (int)(p & 15);
  int  lane= (int)((p >> 4) & 31);
  int  kk  = (int)((p >> 9) & 3);
  long tile= p >> 11;
  long row = tile * 16 + (lane & 15);
  int  k0  = kk * 32 + (lane >> 4) * 8;
  int  k   = k0 + ((j < 8) ? j : (8 + j));
  const float* s = X + row * 128 + k;
  float x0 = s[0], x1 = s[1];
  if (do_relu) { x0 = fmaxf(x0, 0.0f); x1 = fmaxf(x1, 0.0f); }
  Ap2[t] = cvt2(x0, x1);
}

// ---------------------------------------------------------------------------
// Y[N x 128] = A @ W with pre-packed bf16 fragments.
// Block = 256 threads = 8 waves; wave w computes the 16x16 tile at
// rows [16*blockIdx.x, +16), cols [16*w, +16). Inner loop: 2 loads + 1 WMMA.
// ---------------------------------------------------------------------------
__global__ void __launch_bounds__(256)
gemm_bf16_wmma_k(const unsigned short* __restrict__ Apk,
                 const unsigned short* __restrict__ Wp,
                 float* __restrict__ Y) {
  const int  lane = threadIdx.x & 31;
  const int  wave = threadIdx.x >> 5;        // n-tile 0..7
  const long tile = blockIdx.x;
  const unsigned short* ap = Apk + ((tile * 4) * 32 + lane) * 16;
  const unsigned short* bp = Wp  + ((long)(wave * 4) * 32 + lane) * 16;

  v8f c = {};
#pragma unroll
  for (int kk = 0; kk < 4; ++kk) {
    Frag16 a, b;
    a.u = *(const vus16*)(ap + kk * 512);    // 32 bytes, contiguous
    b.u = *(const vus16*)(bp + kk * 512);
    c = __builtin_amdgcn_wmma_f32_16x16x32_bf16(false, a.b, false, b.b,
                                                (short)0, c, false, false);
  }

  const int half = lane >> 4, l16 = lane & 15;
  const int col  = wave * 16 + l16;
  float* y = Y + (tile * 16 + half * 8) * 128 + col;  // lanes 0-15: M=r ; 16-31: M=r+8
#pragma unroll
  for (int r = 0; r < 8; ++r) y[(long)r * 128] = c[r];
}

// ---------------------------------------------------------------------------
// acc[i][f] = bias[f] + dis[i]^2 * src[i][f]      (self-loop message fused)
// ---------------------------------------------------------------------------
__global__ void init_acc_k(const float* __restrict__ src,
                           const float* __restrict__ bias,
                           const float* __restrict__ dis,
                           float* __restrict__ acc) {
  long i = blockIdx.x;
  int  f = threadIdx.x;
  float d = dis[i];
  acc[i * 128 + f] = bias[f] + d * d * src[i * 128 + f];
}

// ---------------------------------------------------------------------------
// Edge scatter: acc[col[e]] += src[row[e]] * (dis[row]*ew*dis[col])
// One block (128 threads) per edge.
// ---------------------------------------------------------------------------
__global__ void scatter_k(const float* __restrict__ src,
                          const int* __restrict__ row,
                          const int* __restrict__ col,
                          const float* __restrict__ ew,
                          const float* __restrict__ dis,
                          float* __restrict__ acc) {
  long e = blockIdx.x;
  int  f = threadIdx.x;
  int  r = row[e], c = col[e];
  float nrm = dis[r] * ew[e] * dis[c];
  atomicAdd(&acc[(long)c * 128 + f], src[(long)r * 128 + f] * nrm);
}

__global__ void zero_k(float* p, int n) {
  for (int i = threadIdx.x; i < n; i += blockDim.x) p[i] = 0.0f;
}

__global__ void pool_k(const float* __restrict__ h,
                       const int* __restrict__ batch,
                       float* __restrict__ pool, float* __restrict__ cnt) {
  long i = blockIdx.x;
  int  f = threadIdx.x;
  int  g = batch[i];
  atomicAdd(&pool[g * 128 + f], h[i * 128 + f]);
  if (f == 0) atomicAdd(&cnt[g], 1.0f);
}

// ---------------------------------------------------------------------------
// Head: pooled = sums/cnt ; z = relu(pooled@Wc1+bc1) ; out = z@Wc2+bc2
// Single block, G=10, H=128, H/2=64, C=10.
// ---------------------------------------------------------------------------
__global__ void classifier_k(const float* __restrict__ pool_sums,
                             const float* __restrict__ cnts,
                             const float* __restrict__ Wc1,
                             const float* __restrict__ bc1,
                             const float* __restrict__ Wc2,
                             const float* __restrict__ bc2,
                             float* __restrict__ out) {
  __shared__ float pooled[10 * 128];
  __shared__ float z[10 * 64];
  int t = threadIdx.x;
  for (int i = t; i < 10 * 128; i += blockDim.x) {
    int g = i >> 7;
    float c = cnts[g];
    pooled[i] = pool_sums[i] / fmaxf(c, 1.0f);
  }
  __syncthreads();
  for (int i = t; i < 10 * 64; i += blockDim.x) {
    int g = i >> 6, j = i & 63;
    float s = bc1[j];
    for (int k = 0; k < 128; ++k) s += pooled[g * 128 + k] * Wc1[k * 64 + j];
    z[i] = fmaxf(s, 0.0f);
  }
  __syncthreads();
  for (int i = t; i < 100; i += blockDim.x) {
    int g = i / 10, c = i % 10;
    float s = bc2[c];
    for (int j = 0; j < 64; ++j) s += z[g * 64 + j] * Wc2[j * 10 + c];
    out[i] = s;
  }
}

// ---------------------------------------------------------------------------
extern "C" void kernel_launch(void* const* d_in, const int* in_sizes, int n_in,
                              void* d_out, int out_size, void* d_ws, size_t ws_size,
                              hipStream_t stream) {
  (void)n_in; (void)out_size; (void)ws_size;

  const float* x    = (const float*)d_in[0];
  const int*   eidx = (const int*)  d_in[1];
  const float* ew   = (const float*)d_in[2];
  const int*   bat  = (const int*)  d_in[3];
  const float* W1   = (const float*)d_in[4];
  const float* b1   = (const float*)d_in[5];
  const float* W2   = (const float*)d_in[6];
  const float* b2   = (const float*)d_in[7];
  const float* Wc1  = (const float*)d_in[8];
  const float* bc1  = (const float*)d_in[9];
  const float* Wc2  = (const float*)d_in[10];
  const float* bc2  = (const float*)d_in[11];
  float* out = (float*)d_out;

  const int N = in_sizes[0] / 128;       // 50000 (divisible by 16)
  const int E = in_sizes[2];             // 800000
  const int* row = eidx;
  const int* col = eidx + E;

  // --- workspace carve-up (256B aligned) ---
  char* ws = (char*)d_ws;
  size_t o = 0;
  auto carve = [&](size_t bytes) -> char* {
    char* p = ws + o;
    o += (bytes + 255) & ~(size_t)255;
    return p;
  };
  float*          deg  = (float*)         carve((size_t)N * 4);
  float*          dis  = (float*)         carve((size_t)N * 4);
  unsigned short* Wp1  = (unsigned short*)carve(16384 * 2);
  unsigned short* Wp2  = (unsigned short*)carve(16384 * 2);
  unsigned short* Apk  = (unsigned short*)carve((size_t)N * 128 * 2);
  float*          bufA = (float*)         carve((size_t)N * 128 * 4);
  float*          bufB = (float*)         carve((size_t)N * 128 * 4);
  float*          pool = (float*)         carve(10 * 128 * 4);
  float*          cnt  = (float*)         carve(10 * 4);

  // --- normalization ---
  init_deg_k <<<(N + 255) / 256, 256, 0, stream>>>(deg, N);
  deg_accum_k<<<(E + 255) / 256, 256, 0, stream>>>(col, ew, deg, E);
  dis_k      <<<(N + 255) / 256, 256, 0, stream>>>(deg, dis, N);

  // --- weight packing (bf16 fragment order) ---
  pack_w_k<<<64, 256, 0, stream>>>(W1, Wp1);
  pack_w_k<<<64, 256, 0, stream>>>(W2, Wp2);

  const int  gemm_blocks = N / 16;             // 3125 (exact)
  const long npairs      = (long)N * 64;       // bf16 pairs per activation matrix
  const int  pack_blocks = (int)((npairs + 255) / 256);

  // --- conv 1 ---
  pack_a_k<<<pack_blocks, 256, 0, stream>>>(x, (unsigned*)Apk, 0, npairs);
  gemm_bf16_wmma_k<<<gemm_blocks, 256, 0, stream>>>(Apk, Wp1, bufA);
  init_acc_k<<<N, 128, 0, stream>>>(bufA, b1, dis, bufB);
  scatter_k <<<E, 128, 0, stream>>>(bufA, row, col, ew, dis, bufB);

  // --- conv 2 (ReLU fused into the A-pack) ---
  pack_a_k<<<pack_blocks, 256, 0, stream>>>(bufB, (unsigned*)Apk, 1, npairs);
  gemm_bf16_wmma_k<<<gemm_blocks, 256, 0, stream>>>(Apk, Wp2, bufA);
  init_acc_k<<<N, 128, 0, stream>>>(bufA, b2, dis, bufB);
  scatter_k <<<E, 128, 0, stream>>>(bufA, row, col, ew, dis, bufB);

  // --- pool + head ---
  zero_k<<<1, 256, 0, stream>>>(pool, 10 * 128);
  zero_k<<<1, 256, 0, stream>>>(cnt, 10);
  pool_k<<<N, 128, 0, stream>>>(bufB, bat, pool, cnt);
  classifier_k<<<1, 128, 0, stream>>>(pool, cnt, Wc1, bc1, Wc2, bc2, out);
}
